// LSTMModel_16381005266960
// MI455X (gfx1250) — matrix operand order repaired and checked by reading, not compiled, not run on hardware
//
#include <hip/hip_runtime.h>
#include <hip/hip_bf16.h>

// ---------- types for WMMA ----------
typedef __attribute__((ext_vector_type(16))) __bf16 v16bf;
typedef __attribute__((ext_vector_type(8)))  float  v8f;

struct __align__(16) U4 { unsigned int a, b, c, d; };
struct __align__(16) F4 { float x, y, z, w; };

__device__ __forceinline__ unsigned short f2bf(float f) {
  unsigned int u = __float_as_uint(f);
  unsigned int r = 0x7FFFu + ((u >> 16) & 1u);   // round-to-nearest-even
  return (unsigned short)((u + r) >> 16);
}
__device__ __forceinline__ float bf2f(unsigned short s) {
  return __uint_as_float(((unsigned int)s) << 16);
}
__device__ __forceinline__ float sigmoidf_(float v) {
  return 1.0f / (1.0f + __expf(-v));
}

// build a v16bf fragment from two 16-byte LDS chunks (8 + 8 bf16)
__device__ __forceinline__ v16bf make_frag(const unsigned short* p0, const unsigned short* p1) {
  struct __align__(32) P { U4 a, b; } t;
  t.a = *(const U4*)p0;
  t.b = *(const U4*)p1;
  return __builtin_bit_cast(v16bf, t);
}

// ---------- CDNA5 async global->LDS copy (bypasses VGPRs, tracked by ASYNCcnt) ----------
// GVS addressing: mem = SGPR64 base + per-lane u32 byte offset. VDST = per-lane LDS byte addr.
__device__ __forceinline__ void async_b128(unsigned lds_addr, unsigned byte_off,
                                           unsigned long long base) {
  asm volatile("global_load_async_to_lds_b128 %0, %1, %2"
               :: "v"(lds_addr), "v"(byte_off), "s"(base)
               : "memory");
}
__device__ __forceinline__ void wait_async0() {
#if __has_builtin(__builtin_amdgcn_s_wait_asynccnt)
  __builtin_amdgcn_s_wait_asynccnt(0);
#else
  asm volatile("s_wait_asynccnt 0x0" ::: "memory");
#endif
}
// generic shared pointer -> LDS byte address (low 32 bits of the shared aperture address)
__device__ __forceinline__ unsigned lds_addr_of(const void* p) {
  return (unsigned)(unsigned long long)p;
}

// ---------- constants ----------
// B=256, T=500, I=256, H=512, 4H=2048, O=100
#define LSTM_T 500
#define LSTM_I 256
#define LSTM_H 512
#define LSTM_G 2048
#define LSTM_O 100
#define LDA 40   // padded LDS row stride in bf16 elements (80B: 16B aligned, conflict-breaking)

// ---------- init: convert weights to bf16, fuse biases, zero state ----------
__global__ void lstm_init(const float* __restrict__ Wx, const float* __restrict__ bx,
                          const float* __restrict__ Wh, const float* __restrict__ bh,
                          unsigned short* __restrict__ Wxb, unsigned short* __restrict__ Whb,
                          float* __restrict__ cbuf, unsigned short* __restrict__ h0,
                          unsigned short* __restrict__ h1, float* __restrict__ bias) {
  const int idx = blockIdx.x * blockDim.x + threadIdx.x;   // 0 .. 1048575
  if (idx < LSTM_G * LSTM_I) Wxb[idx] = f2bf(Wx[idx]);     // 524288
  if (idx < LSTM_G * LSTM_H) Whb[idx] = f2bf(Wh[idx]);     // 1048576
  if (idx < 256 * LSTM_H) { cbuf[idx] = 0.0f; h0[idx] = 0; h1[idx] = 0; }
  if (idx < LSTM_G) bias[idx] = bx[idx] + bh[idx];
}

// ---------- one LSTM timestep ----------
// Block: 256 threads (8 waves). Tile: 128 batch rows x 16 hidden cols.
// Each wave owns 16 rows and accumulates 4 gate tiles (i,f,g,o) over K = 256(x) + 512(h).
// Double-buffered LDS pipeline: async-stage chunk k+1 while WMMAs consume chunk k.
__global__ __launch_bounds__(256)
void lstm_step(const float* __restrict__ x, int t,
               const unsigned short* __restrict__ Wxb,
               const unsigned short* __restrict__ Whb,
               const float* __restrict__ bias,
               const unsigned short* __restrict__ hin,
               unsigned short* __restrict__ hout,
               float* __restrict__ cbuf) {
  __shared__ __align__(16) unsigned short lA[2][128 * LDA];  // 2 x 10 KB activations
  __shared__ __align__(16) unsigned short lB[2][64 * LDA];   // 2 x  5 KB weights (4 gate tiles)

  const int tid   = threadIdx.x;
  const int wave  = tid >> 5;
  const int lane  = tid & 31;
  const int bRow0 = blockIdx.x * 128;        // batch-row base (grid.x = 2)
  const int n0    = blockIdx.y * 16;         // hidden-col base (grid.y = 32)

  v8f acc0 = {}, acc1 = {}, acc2 = {}, acc3 = {};

  // per-lane fragment addressing (contiguous k = 16*(lane/16) + s, same order for A and B, wave32)
  const int mrow = wave * 16 + (lane & 15);  // A: M = lane%16 within wave tile
  const int kA0  = (lane < 16) ? 0 : 16;     // A 16-bit: K {0-15} / {16-31}
  const int nB   = lane & 15;                // B: N = lane%16
  const int kB0  = (lane < 16) ? 0 : 16;     // B 16-bit: K {0-15} / {16-31}

  // stage K-chunk kc into LDS buffer `buf`
  auto stage = [&](int kc, int buf) {
    const int kk = kc * 32;
    // ---- B tile: 4 gates x 16 n x 32 k bf16 = one b128 per thread, async ----
    {
      const int gate = tid >> 6;
      const int n    = (tid >> 2) & 15;
      const int ko   = (tid & 3) << 3;
      const unsigned ldsb = lds_addr_of(&lB[buf][(gate * 16 + n) * LDA + ko]);
      if (kk < LSTM_I) {
        const unsigned off = (unsigned)(((gate * LSTM_H + n0 + n) * LSTM_I + kk + ko) * 2);
        async_b128(ldsb, off, (unsigned long long)Wxb);
      } else {
        const unsigned off =
            (unsigned)(((gate * LSTM_H + n0 + n) * LSTM_H + (kk - LSTM_I) + ko) * 2);
        async_b128(ldsb, off, (unsigned long long)Whb);
      }
    }
    // ---- A tile ----
    if (kk < LSTM_I) {
      // from x[b, t, kk..kk+31]: needs f32->bf16 conversion -> manual path
      for (int i = tid; i < 1024; i += 256) {            // 128 rows x 8 quads
        const int row = i >> 3;
        const int kq  = (i & 7) << 2;
        const F4 v = *(const F4*)(x + ((size_t)(bRow0 + row) * LSTM_T + t) * LSTM_I + kk + kq);
        unsigned short* dst = &lA[buf][row * LDA + kq];
        dst[0] = f2bf(v.x); dst[1] = f2bf(v.y); dst[2] = f2bf(v.z); dst[3] = f2bf(v.w);
      }
    } else {
      // from h (already bf16): async copy, 128 rows x 4 oct-vectors = 2 per thread
      const int kh = kk - LSTM_I;
      for (int i = tid; i < 512; i += 256) {
        const int row = i >> 2;
        const int ko  = (i & 3) << 3;
        const unsigned ldsb = lds_addr_of(&lA[buf][row * LDA + ko]);
        const unsigned off  = (unsigned)((((bRow0 + row) * LSTM_H) + kh + ko) * 2);
        async_b128(ldsb, off, (unsigned long long)hin);
      }
    }
  };

  stage(0, 0);
  wait_async0();
  __syncthreads();

  for (int kc = 0; kc < 24; ++kc) {          // K chunks of 32: 8 from x, 16 from h
    const int buf = kc & 1;
    if (kc + 1 < 24) stage(kc + 1, buf ^ 1); // overlap next chunk's copy with WMMAs

    const unsigned short* arow = &lA[buf][mrow * LDA + kA0];
    const v16bf a = make_frag(arow, arow + 8);
    const unsigned short* bb = &lB[buf][nB * LDA + kB0];
    const v16bf b0 = make_frag(bb,            bb + 8);
    const v16bf b1 = make_frag(bb + 16 * LDA, bb + 16 * LDA + 8);
    const v16bf b2 = make_frag(bb + 32 * LDA, bb + 32 * LDA + 8);
    const v16bf b3 = make_frag(bb + 48 * LDA, bb + 48 * LDA + 8);

    acc0 = __builtin_amdgcn_wmma_f32_16x16x32_bf16(false, a, false, b0, (short)0, acc0, false, false);
    acc1 = __builtin_amdgcn_wmma_f32_16x16x32_bf16(false, a, false, b1, (short)0, acc1, false, false);
    acc2 = __builtin_amdgcn_wmma_f32_16x16x32_bf16(false, a, false, b2, (short)0, acc2, false, false);
    acc3 = __builtin_amdgcn_wmma_f32_16x16x32_bf16(false, a, false, b3, (short)0, acc3, false, false);

    wait_async0();       // next buffer fully in LDS (this wave's copies)
    __syncthreads();     // ... and everyone else's; also: all reads of `buf` done
  }

  // ---- gate nonlinearities + c/h update ----
  // C/D layout: VGPR r, lanes 0-15 -> M=r, N=lane; lanes 16-31 -> M=8+r, N=lane-16
  const int col   = n0 + nB;
  const int mBase = (lane < 16) ? 0 : 8;
  const float bI = bias[col];
  const float bF = bias[LSTM_H + col];
  const float bG = bias[2 * LSTM_H + col];
  const float bO = bias[3 * LSTM_H + col];

#pragma unroll
  for (int r = 0; r < 8; ++r) {
    const int bRow = bRow0 + wave * 16 + mBase + r;
    const size_t idx = (size_t)bRow * LSTM_H + col;
    const float gi = sigmoidf_(acc0[r] + bI);
    const float gf = sigmoidf_(acc1[r] + bF);
    const float gg = tanhf(acc2[r] + bG);
    const float go = sigmoidf_(acc3[r] + bO);
    const float cN = cbuf[idx] * gf + gi * gg;
    cbuf[idx] = cN;
    hout[idx] = f2bf(go * tanhf(cN));
  }
}

// ---------- final FC: out[b,o] = h[b,:] . Wfc[o,:] + bfc[o] ----------
__global__ void lstm_fc(const unsigned short* __restrict__ hfin,
                        const float* __restrict__ Wfc, const float* __restrict__ bfc,
                        float* __restrict__ out) {
  const int o = blockIdx.x;      // 0..99
  const int b = threadIdx.x;     // 0..255
  const float* w = Wfc + (size_t)o * LSTM_H;
  const unsigned short* hv = hfin + (size_t)b * LSTM_H;
  float s = 0.0f;
  for (int k = 0; k < LSTM_H; ++k) s += bf2f(hv[k]) * w[k];
  out[(size_t)b * LSTM_O + o] = s + bfc[o];
}

extern "C" void kernel_launch(void* const* d_in, const int* in_sizes, int n_in,
                              void* d_out, int out_size, void* d_ws, size_t ws_size,
                              hipStream_t stream) {
  (void)in_sizes; (void)n_in; (void)out_size; (void)ws_size;
  const float* x   = (const float*)d_in[0];
  const float* Wx  = (const float*)d_in[1];
  const float* bx  = (const float*)d_in[2];
  const float* Wh  = (const float*)d_in[3];
  const float* bh  = (const float*)d_in[4];
  const float* Wfc = (const float*)d_in[5];
  const float* bfc = (const float*)d_in[6];
  float* out = (float*)d_out;

  // workspace layout (~4.2 MB)
  unsigned short* Wxb = (unsigned short*)d_ws;          // 524288 bf16
  unsigned short* Whb = Wxb + (size_t)LSTM_G * LSTM_I;  // 1048576 bf16
  float* cbuf = (float*)(Whb + (size_t)LSTM_G * LSTM_H);// 131072 f32
  unsigned short* h0 = (unsigned short*)(cbuf + 256 * LSTM_H); // 131072 bf16
  unsigned short* h1 = h0 + 256 * LSTM_H;                      // 131072 bf16
  float* bias = (float*)(h1 + 256 * LSTM_H);                   // 2048 f32

  lstm_init<<<4096, 256, 0, stream>>>(Wx, bx, Wh, bh, Wxb, Whb, cbuf, h0, h1, bias);

  unsigned short* hb[2] = { h0, h1 };
  for (int t = 0; t < LSTM_T; ++t) {
    lstm_step<<<dim3(2, 32), 256, 0, stream>>>(x, t, Wxb, Whb, bias,
                                               hb[t & 1], hb[(t + 1) & 1], cbuf);
  }
  // after step t=499 the final h lives in hb[(499+1)&1] == hb[0]
  lstm_fc<<<LSTM_O, 256, 0, stream>>>(hb[0], Wfc, bfc, out);
}
